// FiLMedGNN_86801289052891
// MI455X (gfx1250) — matrix operand order repaired and checked by reading, not compiled, not run on hardware
//
#include <hip/hip_runtime.h>

// ---------------- problem constants (from reference) ----------------
#define NN   20000   // nodes
#define EE   640000  // edges
#define BB   1024    // batch
#define FIN  14615   // node feature dim
#define HID  128     // hidden
#define CIN  29
#define CHID 256
#define KCHUNK 512   // K-split chunk for hT@Wproj (20000/512 -> 40 WGs)
#define NT1  ((FIN + 31) / 32)   // 457 K-tiles for x@W1
#define NT2  (NN / 32)           // 625 K-tiles for hT@Wproj (exact)
#define TILE_US 4096             // one 32(K) x 128(N) bf16 tile, stored [n][k]

typedef __attribute__((ext_vector_type(16))) __bf16         v16bf;
typedef __attribute__((ext_vector_type(16))) unsigned short v16us;
typedef __attribute__((ext_vector_type(8)))  unsigned short v8us;
typedef __attribute__((ext_vector_type(8)))  float          v8f;
typedef int v4i_g __attribute__((vector_size(16)));   // matches builtin param pointee

// fp32 -> bf16 round-to-nearest-even (bit pattern)
__device__ __forceinline__ unsigned short f2bf(float f) {
  unsigned int u = __builtin_bit_cast(unsigned int, f);
  unsigned int r = u + 0x7FFFu + ((u >> 16) & 1u);
  return (unsigned short)(r >> 16);
}

// ---------------- CDNA5 async global->LDS copy (ASYNCcnt) ----------------
#if defined(__has_builtin)
#if __has_builtin(__builtin_amdgcn_global_load_async_to_lds_b128) && \
    __has_builtin(__builtin_amdgcn_s_wait_asynccnt)
#define USE_ASYNC_LDS 1
#endif
#endif

#ifdef USE_ASYNC_LDS
__device__ __forceinline__ void cp16_async(const unsigned short* g, unsigned short* l) {
  __builtin_amdgcn_global_load_async_to_lds_b128(
      (__attribute__((address_space(1))) v4i_g*)(g),
      (__attribute__((address_space(3))) v4i_g*)(l), 0, 0);
}
#endif

__device__ __forceinline__ void wait_async_all() {
#ifdef USE_ASYNC_LDS
  __builtin_amdgcn_s_wait_asynccnt(0);
#endif
}

// Copy one contiguous 8KB tile (4096 bf16, already [n][k]) global -> LDS.
// 256 threads; thread t moves 32 bytes as 2x b128.
__device__ __forceinline__ void stage_tile(const unsigned short* __restrict__ g,
                                           unsigned short* __restrict__ l) {
  const int t = threadIdx.x;
#ifdef USE_ASYNC_LDS
  cp16_async(g + t * 16,     l + t * 16);
  cp16_async(g + t * 16 + 8, l + t * 16 + 8);
#else
  *(v8us*)(l + t * 16)     = *(const v8us*)(g + t * 16);
  *(v8us*)(l + t * 16 + 8) = *(const v8us*)(g + t * 16 + 8);
#endif
}

// A-matrix fragment (16x32 bf16) from [n][k] tile:
// lanes 0-15 hold K 0-7 / 16-23, lanes 16-31 hold K 8-15 / 24-31
__device__ __forceinline__ v16bf fragA_ldsT(const unsigned short* __restrict__ lds,
                                            int col, int lane) {
  const int kb = (lane < 16) ? 0 : 8;
  v8us lo = *(const v8us*)(lds + col * 32 + kb);
  v8us hi = *(const v8us*)(lds + col * 32 + kb + 16);
  v16us v;
#pragma unroll
  for (int i = 0; i < 8; i++) { v[i] = lo[i]; v[8 + i] = hi[i]; }
  return __builtin_bit_cast(v16bf, v);
}

// B-matrix fragment (32x16 bf16) from [n][k] tile:
// lanes 0-15 hold K 0-7 / 16-23, lanes 16-31 hold K 8-15 / 24-31
__device__ __forceinline__ v16bf fragB_ldsT(const unsigned short* __restrict__ lds,
                                            int col, int lane) {
  const int kb = (lane < 16) ? 0 : 8;
  v8us lo = *(const v8us*)(lds + col * 32 + kb);
  v8us hi = *(const v8us*)(lds + col * 32 + kb + 16);
  v16us v;
#pragma unroll
  for (int i = 0; i < 8; i++) { v[i] = lo[i]; v[8 + i] = hi[i]; }
  return __builtin_bit_cast(v16bf, v);
}

// -------------------------------------------------------------------
// W1 -> bf16, tile-transposed: w1t[kt*4096 + n*32 + k] = bf16(W1[(kt*32+k)*128 + n])
__global__ void k_w1_tiles(const float* __restrict__ W1, unsigned short* __restrict__ w1t) {
  int i = blockIdx.x * blockDim.x + threadIdx.x;
  if (i < NT1 * TILE_US) {
    int kt = i >> 12, r = i & 4095;
    int n = r >> 5, k = r & 31;
    int kg = kt * 32 + k;
    w1t[i] = (kg < FIN) ? f2bf(W1[(size_t)kg * HID + n]) : (unsigned short)0;
  }
}

// Wproj -> bf16, tile-transposed (20000 % 32 == 0, no tail)
__global__ void k_wp_tiles(const float* __restrict__ Wp, unsigned short* __restrict__ wpt) {
  int i = blockIdx.x * blockDim.x + threadIdx.x;
  if (i < NT2 * TILE_US) {
    int kt = i >> 12, r = i & 4095;
    int n = r >> 5, k = r & 31;
    wpt[i] = f2bf(Wp[(size_t)(kt * 32 + k) * HID + n]);
  }
}

// deg = 1 (self loop), Mmat = 0
__global__ void k_init(float* __restrict__ deg, float* __restrict__ Mmat) {
  int i = blockIdx.x * blockDim.x + threadIdx.x;
  if (i < NN) deg[i] = 1.0f;
  if (i < HID * HID) Mmat[i] = 0.0f;
}

// cond = condition @ Wm + bm  [1024,256], K=29
__global__ __launch_bounds__(256) void k_cond(const float* __restrict__ condition,
                                              const float* __restrict__ Wm,
                                              const float* __restrict__ bm,
                                              float* __restrict__ cond) {
  __shared__ float sc[CIN];
  const int b = blockIdx.x, n = threadIdx.x;
  if (n < CIN) sc[n] = condition[b * CIN + n];
  __syncthreads();
  float acc = bm[n];
#pragma unroll
  for (int k = 0; k < CIN; k++) acc += sc[k] * Wm[k * CHID + n];
  cond[b * CHID + n] = acc;
}

// gamma = cond@Wg + bg, beta = cond@Wb + bb  [1024,128], K=256
__global__ __launch_bounds__(128) void k_gamma_beta(const float* __restrict__ cond,
                                                    const float* __restrict__ Wg,
                                                    const float* __restrict__ bg,
                                                    const float* __restrict__ Wb,
                                                    const float* __restrict__ bb,
                                                    float* __restrict__ gamma,
                                                    float* __restrict__ beta) {
  __shared__ float sc[CHID];
  const int b = blockIdx.x, j = threadIdx.x;
  sc[j] = cond[b * CHID + j];
  sc[j + 128] = cond[b * CHID + 128 + j];
  __syncthreads();
  float g = bg[j], be = bb[j];
  for (int k = 0; k < CHID; k++) {
    float c = sc[k];
    g  += c * Wg[k * HID + j];
    be += c * Wb[k * HID + j];
  }
  gamma[b * HID + j] = g;
  beta[b * HID + j]  = be;
}

// h0 = x @ W1 (bf16 WMMA, fp32 acc). Streams x once (memory-bound term).
// 256 thr = 8 waves; wave w owns rows [blk*128 + 16w, +16) x all 128 cols.
// W1 tiles double-buffered in LDS via async global->LDS copies.
__global__ __launch_bounds__(256) void k_gemm_xw1(const float* __restrict__ x,
                                                  const unsigned short* __restrict__ w1t,
                                                  float* __restrict__ h0) {
  __shared__ unsigned short ldsb[2][TILE_US];
  const int t = threadIdx.x, lane = t & 31, wave = t >> 5;
  const int mbase = blockIdx.x * 128 + wave * 16;
  const int ml = lane & 15;
  const int kb = (lane < 16) ? 0 : 8;
  int rowL = mbase + ml;
  if (rowL >= NN) rowL = NN - 1;   // clamp loads; stores guarded below
  const float* __restrict__ xrow = x + (size_t)rowL * FIN;

  v8f acc[8];
#pragma unroll
  for (int i = 0; i < 8; i++)
#pragma unroll
    for (int j = 0; j < 8; j++) acc[i][j] = 0.0f;

  stage_tile(w1t, ldsb[0]);   // prefetch first tile

  for (int kt = 0; kt < NT1; kt++) {
    const int cur = kt & 1;
    const int k0 = kt * 32;
    wait_async_all();          // our copies for 'cur' done
    __syncthreads();           // everyone's copies done; prev reads of cur^1 done
    if (kt + 1 < NT1) stage_tile(w1t + (size_t)(kt + 1) * TILE_US, ldsb[cur ^ 1]);
    __builtin_prefetch(xrow + k0 + 160, 0, 0);   // global_prefetch_b8 on x stream

    // A fragment: convert fp32 row data to bf16 in registers
    v16us av;
    if (k0 + 32 <= FIN) {
#pragma unroll
      for (int i = 0; i < 8; i++) av[i] = f2bf(xrow[k0 + kb + i]);
#pragma unroll
      for (int i = 0; i < 8; i++) av[8 + i] = f2bf(xrow[k0 + kb + 16 + i]);
    } else {
#pragma unroll
      for (int i = 0; i < 16; i++) {
        int kk = k0 + kb + (i < 8 ? i : 8 + i);
        av[i] = (kk < FIN) ? f2bf(xrow[kk]) : (unsigned short)0;
      }
    }
    v16bf a = __builtin_bit_cast(v16bf, av);

#pragma unroll
    for (int nt = 0; nt < 8; nt++) {
      v16bf b = fragB_ldsT(ldsb[cur], nt * 16 + ml, lane);
      acc[nt] = __builtin_amdgcn_wmma_f32_16x16x32_bf16(
          false, a, false, b, (short)0, acc[nt], false, false);
    }
  }

  const int rbase = mbase + ((lane < 16) ? 0 : 8);
#pragma unroll
  for (int nt = 0; nt < 8; nt++) {
    const int n0 = nt * 16 + ml;
#pragma unroll
    for (int v = 0; v < 8; v++) {
      int r = rbase + v;
      if (r < NN) h0[(size_t)r * HID + n0] = acc[nt][v];
    }
  }
}

// deg scatter: deg[dst] += 1
__global__ void k_deg(const int* __restrict__ ei, float* __restrict__ deg) {
  int e = blockIdx.x * blockDim.x + threadIdx.x;
  if (e < EE) atomicAdd(&deg[ei[EE + e]], 1.0f);
}

__global__ void k_dinv(const float* __restrict__ deg, float* __restrict__ dinv) {
  int i = blockIdx.x * blockDim.x + threadIdx.x;
  if (i < NN) { float d = deg[i]; dinv[i] = (d > 0.0f) ? rsqrtf(d) : 0.0f; }
}

// agg = dinv[i]^2 * h0[i] + b1  (self-loop term + bias)
__global__ void k_agginit(const float* __restrict__ h0, const float* __restrict__ dinv,
                          const float* __restrict__ b1, float* __restrict__ agg) {
  int i = blockIdx.x * blockDim.x + threadIdx.x;
  if (i < NN * HID) {
    int node = i >> 7, c = i & 127;
    float w = dinv[node];
    agg[i] = w * w * h0[i] + b1[c];
  }
}

// agg[dst] += dinv[src]*dinv[dst] * h0[src]; one thread per (edge, 4-col chunk).
// agg is 10 MB -> resident in the 192MB L2, so f32 atomics run at L2 rate.
__global__ void k_aggscatter(const int* __restrict__ ei, const float* __restrict__ dinv,
                             const float* __restrict__ h0, float* __restrict__ agg) {
  int g = blockIdx.x * blockDim.x + threadIdx.x;
  int e = g >> 5;
  int q = (g & 31) * 4;
  if (e < EE) {
    int s = ei[e], d = ei[EE + e];
    float w = dinv[s] * dinv[d];
    const float* hs = h0 + (size_t)s * HID + q;
    float* ad = agg + (size_t)d * HID + q;
#pragma unroll
    for (int c = 0; c < 4; c++) atomicAdd(ad + c, w * hs[c]);
  }
}

// hbf_t = bf16(relu(agg)), written tile-transposed [kt][n][k] for GEMM2 staging
__global__ void k_relubf_t(const float* __restrict__ agg, unsigned short* __restrict__ hbt) {
  int i = blockIdx.x * blockDim.x + threadIdx.x;
  if (i < NN * HID) {
    int node = i >> 7, c = i & 127;
    int kt = node >> 5, k = node & 31;
    hbt[(size_t)kt * TILE_US + c * 32 + k] = f2bf(fmaxf(agg[i], 0.0f));
  }
}

// Mmat += h^T @ Wproj over a K-chunk of nodes (bf16 WMMA, K-split + f32 atomics).
// Replaces (gamma @ h^T) @ Wproj == gamma @ (h^T @ Wproj): kills 82MB intermediate.
__global__ __launch_bounds__(256) void k_gemm_htwp(const unsigned short* __restrict__ hbt,
                                                   const unsigned short* __restrict__ wpt,
                                                   float* __restrict__ Mmat) {
  __shared__ unsigned short lds_a[2][TILE_US];
  __shared__ unsigned short lds_b[2][TILE_US];
  const int t = threadIdx.x, lane = t & 31, wave = t >> 5;
  const int ml = lane & 15;
  const int t0 = blockIdx.x * (KCHUNK / 32);
  int ntl = NT2 - t0; if (ntl > KCHUNK / 32) ntl = KCHUNK / 32;

  v8f acc[8];
#pragma unroll
  for (int i = 0; i < 8; i++)
#pragma unroll
    for (int j = 0; j < 8; j++) acc[i][j] = 0.0f;

  stage_tile(hbt + (size_t)t0 * TILE_US, lds_a[0]);
  stage_tile(wpt + (size_t)t0 * TILE_US, lds_b[0]);

  for (int i = 0; i < ntl; i++) {
    const int cur = i & 1;
    wait_async_all();
    __syncthreads();
    if (i + 1 < ntl) {
      stage_tile(hbt + (size_t)(t0 + i + 1) * TILE_US, lds_a[cur ^ 1]);
      stage_tile(wpt + (size_t)(t0 + i + 1) * TILE_US, lds_b[cur ^ 1]);
    }
    v16bf a = fragA_ldsT(lds_a[cur], wave * 16 + ml, lane);
#pragma unroll
    for (int nt = 0; nt < 8; nt++) {
      v16bf b = fragB_ldsT(lds_b[cur], nt * 16 + ml, lane);
      acc[nt] = __builtin_amdgcn_wmma_f32_16x16x32_bf16(
          false, a, false, b, (short)0, acc[nt], false, false);
    }
  }

  const int rbase = wave * 16 + ((lane < 16) ? 0 : 8);
#pragma unroll
  for (int nt = 0; nt < 8; nt++) {
    const int n0 = nt * 16 + ml;
#pragma unroll
    for (int v = 0; v < 8; v++)
      atomicAdd(&Mmat[(size_t)(rbase + v) * HID + n0], acc[nt][v]);
  }
}

// z = relu(gamma@Mmat + bproj + beta); out = log_softmax(z@Wout + bout)
__global__ __launch_bounds__(128) void k_final(const float* __restrict__ gamma,
                                               const float* __restrict__ beta,
                                               const float* __restrict__ Mmat,
                                               const float* __restrict__ bproj,
                                               const float* __restrict__ Wout,
                                               const float* __restrict__ bout,
                                               float* __restrict__ out) {
  __shared__ float sg[HID];
  __shared__ float red0[HID];
  __shared__ float red1[HID];
  const int b = blockIdx.x, j = threadIdx.x;
  sg[j] = gamma[b * HID + j];
  __syncthreads();
  float z = bproj[j];
  for (int k = 0; k < HID; k++) z += sg[k] * Mmat[k * HID + j];
  z += beta[b * HID + j];
  z = fmaxf(z, 0.0f);
  red0[j] = z * Wout[j * 2 + 0];
  red1[j] = z * Wout[j * 2 + 1];
  __syncthreads();
  for (int s = 64; s > 0; s >>= 1) {
    if (j < s) { red0[j] += red0[j + s]; red1[j] += red1[j + s]; }
    __syncthreads();
  }
  if (j == 0) {
    float o0 = red0[0] + bout[0];
    float o1 = red1[0] + bout[1];
    float mx = fmaxf(o0, o1);
    float lse = mx + logf(expf(o0 - mx) + expf(o1 - mx));
    out[b * 2 + 0] = o0 - lse;
    out[b * 2 + 1] = o1 - lse;
  }
}

// -------------------------------------------------------------------
extern "C" void kernel_launch(void* const* d_in, const int* in_sizes, int n_in,
                              void* d_out, int out_size, void* d_ws, size_t ws_size,
                              hipStream_t stream) {
  const float* x         = (const float*)d_in[0];
  const int*   ei        = (const int*)d_in[1];
  const float* condition = (const float*)d_in[2];
  /* d_in[3] = hidden_dim scalar, == HID */
  const float* Wm    = (const float*)d_in[4];
  const float* bm    = (const float*)d_in[5];
  const float* Wg    = (const float*)d_in[6];
  const float* bg    = (const float*)d_in[7];
  const float* Wb    = (const float*)d_in[8];
  const float* bb    = (const float*)d_in[9];
  const float* W1    = (const float*)d_in[10];
  const float* b1    = (const float*)d_in[11];
  const float* Wproj = (const float*)d_in[12];
  const float* bproj = (const float*)d_in[13];
  const float* Wout  = (const float*)d_in[14];
  const float* bout  = (const float*)d_in[15];
  float* out = (float*)d_out;

  // workspace carve-out (256B aligned)
  char* ws = (char*)d_ws;
  size_t off = 0;
  auto carve = [&](size_t bytes) -> void* {
    void* p = ws + off;
    off += (bytes + 255) & ~(size_t)255;
    return p;
  };
  unsigned short* w1t = (unsigned short*)carve((size_t)NT1 * TILE_US * 2);
  unsigned short* wpt = (unsigned short*)carve((size_t)NT2 * TILE_US * 2);
  unsigned short* hbt = (unsigned short*)carve((size_t)NT2 * TILE_US * 2);
  float* h0    = (float*)carve((size_t)NN * HID * 4);
  float* agg   = (float*)carve((size_t)NN * HID * 4);
  float* deg   = (float*)carve((size_t)NN * 4);
  float* dinv  = (float*)carve((size_t)NN * 4);
  float* cond  = (float*)carve((size_t)BB * CHID * 4);
  float* gamma = (float*)carve((size_t)BB * HID * 4);
  float* beta  = (float*)carve((size_t)BB * HID * 4);
  float* Mmat  = (float*)carve((size_t)HID * HID * 4);

  // weight retiling + init
  k_w1_tiles<<<(NT1 * TILE_US + 255) / 256, 256, 0, stream>>>(W1, w1t);
  k_wp_tiles<<<(NT2 * TILE_US + 255) / 256, 256, 0, stream>>>(Wproj, wpt);
  k_init<<<(NN + 255) / 256, 256, 0, stream>>>(deg, Mmat);

  // condition path
  k_cond<<<BB, 256, 0, stream>>>(condition, Wm, bm, cond);
  k_gamma_beta<<<BB, 128, 0, stream>>>(cond, Wg, bg, Wb, bb, gamma, beta);

  // big GEMM: h0 = x @ W1 (bf16 WMMA, async double-buffered weight tiles)
  k_gemm_xw1<<<(NN + 127) / 128, 256, 0, stream>>>(x, w1t, h0);

  // GCN normalization + aggregation
  k_deg<<<(EE + 255) / 256, 256, 0, stream>>>(ei, deg);
  k_dinv<<<(NN + 255) / 256, 256, 0, stream>>>(deg, dinv);
  k_agginit<<<(NN * HID + 255) / 256, 256, 0, stream>>>(h0, dinv, b1, agg);
  k_aggscatter<<<(EE * 32) / 256, 256, 0, stream>>>(ei, dinv, h0, agg);
  k_relubf_t<<<(NN * HID + 255) / 256, 256, 0, stream>>>(agg, hbt);

  // Mmat = h^T @ Wproj (bf16 WMMA, K-split 40 ways, async double-buffered tiles)
  k_gemm_htwp<<<(NN + KCHUNK - 1) / KCHUNK, 256, 0, stream>>>(hbt, wpt, Mmat);

  // FiLM + out head + log_softmax
  k_final<<<BB, 128, 0, stream>>>(gamma, beta, Mmat, bproj, Wout, bout, out);
}
